// SparseMaskPredictor_26980984553722
// MI455X (gfx1250) — matrix operand order repaired and checked by reading, not compiled, not run on hardware
//
#include <hip/hip_runtime.h>
#include <hip/hip_bf16.h>
#include <math.h>

typedef __attribute__((ext_vector_type(16))) _Float16 v16h;
typedef __attribute__((ext_vector_type(8)))  _Float16 half8;
typedef __attribute__((ext_vector_type(8)))  float    v8f;

#define NROWS 400000
#define DDIM  256
#define HDIM  256

// ---- workspace layout (bytes) ----
#define OFF_W1T   0                        // 256*256 f16 = 131072
#define OFF_W2T   (131072)                 // 128*256 f16 = 65536
#define OFF_W3T   (131072 + 65536)         // 64*128 f16  = 16384
#define OFF_SUMS  (OFF_W3T + 16384)        // 16*128 f32  = 8192
#define OFF_MEANS (OFF_SUMS + 8192)        // 16*128 f16  = 4096
#define OFF_ZLOC  (OFF_MEANS + 4096)       // 400000*128 f16 = 102400000

__device__ __forceinline__ float gelu_exact(float x) {
    return 0.5f * x * (1.0f + erff(x * 0.7071067811865476f));
}

union AFrag { v16h v; half8 h[2]; };

// ---------------- prep: weight transpose/convert to f16, zero seg sums ----------
__global__ void prep_kernel(const float* __restrict__ w1, const float* __restrict__ w2,
                            const float* __restrict__ w3,
                            _Float16* __restrict__ w1T, _Float16* __restrict__ w2T,
                            _Float16* __restrict__ w3T, float* __restrict__ sums) {
    int t = blockIdx.x * blockDim.x + threadIdx.x;
    if (t < 256 * 256) { int n = t >> 8, k = t & 255; w1T[t] = (_Float16)w1[k * 256 + n]; }
    if (t < 128 * 256) { int n = t >> 8, k = t & 255; w2T[t] = (_Float16)w2[k * 128 + n]; }
    if (t <  64 * 128) { int n = t >> 7, k = t & 127; w3T[t] = (_Float16)w3[k *  64 + n]; }
    if (t < 16 * 128)  sums[t] = 0.0f;
}

// ---------------- phase 1: LN -> GEMM1(WMMA f16) -> GELU -> z_local + seg sums ---
__global__ void __launch_bounds__(128, 1)
phase1_kernel(const float* __restrict__ x, const long long* __restrict__ boff,
              const float* __restrict__ lnw, const float* __restrict__ lnb,
              const _Float16* __restrict__ w1T, const float* __restrict__ b1,
              _Float16* __restrict__ zloc, float* __restrict__ sums) {
    __shared__ _Float16 sA[4][16 * 256];   // per-wave LN'ed A tile (f16), 8KB/wave
    __shared__ float    ssum[16 * 128];    // per-block segment sums, 8KB
    __shared__ int      sseg[4][16];       // per-wave segment id of each tile row

    const int tid  = threadIdx.x;
    const int wave = tid >> 5;
    const int lane = tid & 31;
    const bool lo16 = (lane < 16);

    for (int i = tid; i < 16 * 128; i += 128) ssum[i] = 0.0f;
    __syncthreads();

    const int r0 = (blockIdx.x * 4 + wave) * 16;   // 400000 % 64 == 0

    // segment id per tile row: seg(i) = #{j in 1..16 : off[j] <= i}
    if (lo16) {
        long long row = r0 + lane;
        int s = 0;
        #pragma unroll
        for (int j = 1; j <= 16; ++j) s += (boff[j] <= row) ? 1 : 0;
        sseg[wave][lane] = s;
    }

    // ---- LayerNorm: one row per iteration, 8 floats per lane, wave reduction ----
    const int c0 = lane * 8;
    const float4 lw0 = *(const float4*)(lnw + c0);
    const float4 lw1 = *(const float4*)(lnw + c0 + 4);
    const float4 lb0 = *(const float4*)(lnb + c0);
    const float4 lb1 = *(const float4*)(lnb + c0 + 4);
    for (int r = 0; r < 16; ++r) {
        const float* xr = x + (size_t)(r0 + r) * DDIM + c0;
        const float4 p0 = *(const float4*)xr;
        const float4 p1 = *(const float4*)(xr + 4);
        float s  = p0.x + p0.y + p0.z + p0.w + p1.x + p1.y + p1.z + p1.w;
        float ss = p0.x * p0.x + p0.y * p0.y + p0.z * p0.z + p0.w * p0.w +
                   p1.x * p1.x + p1.y * p1.y + p1.z * p1.z + p1.w * p1.w;
        #pragma unroll
        for (int o = 16; o > 0; o >>= 1) {
            s  += __shfl_xor(s,  o, 32);
            ss += __shfl_xor(ss, o, 32);
        }
        const float mu   = s * (1.0f / 256.0f);
        const float var  = ss * (1.0f / 256.0f) - mu * mu;
        const float rstd = rsqrtf(var + 1e-5f);
        half8 hv;
        hv[0] = (_Float16)((p0.x - mu) * rstd * lw0.x + lb0.x);
        hv[1] = (_Float16)((p0.y - mu) * rstd * lw0.y + lb0.y);
        hv[2] = (_Float16)((p0.z - mu) * rstd * lw0.z + lb0.z);
        hv[3] = (_Float16)((p0.w - mu) * rstd * lw0.w + lb0.w);
        hv[4] = (_Float16)((p1.x - mu) * rstd * lw1.x + lb1.x);
        hv[5] = (_Float16)((p1.y - mu) * rstd * lw1.y + lb1.y);
        hv[6] = (_Float16)((p1.z - mu) * rstd * lw1.z + lb1.z);
        hv[7] = (_Float16)((p1.w - mu) * rstd * lw1.w + lb1.w);
        *(half8*)&sA[wave][r * 256 + c0] = hv;
    }

    // ---- hoist A fragments (16x32 f16 layout: half-wave K split 0..15 / 16..31)
    AFrag a[8];
    const int arow = lane & 15;
    const int koff = lo16 ? 0 : 16;
    #pragma unroll
    for (int c = 0; c < 8; ++c) {
        const _Float16* p = &sA[wave][arow * 256 + c * 32 + koff];
        a[c].h[0] = *(const half8*)p;
        a[c].h[1] = *(const half8*)(p + 8);
    }

    // ---- GEMM1: 16 N-tiles x 8 K-chunks of V_WMMA_F32_16X16X32_F16 ----
    // dual accumulators break the D->C RAW chain across the K loop
    const int ncol = lane & 15;
    const int kb2  = lo16 ? 0 : 16;
    const int rb   = lo16 ? 0 : 8;
    for (int n = 0; n < 16; ++n) {
        const int col = n * 16 + ncol;
        const _Float16* bp = w1T + col * 256 + kb2;
        v8f acc0 = {};
        v8f acc1 = {};
        #pragma unroll
        for (int c = 0; c < 8; c += 2) {
            v16h be = *(const v16h*)(bp + c * 32);
            v16h bo = *(const v16h*)(bp + c * 32 + 32);
            acc0 = __builtin_amdgcn_wmma_f32_16x16x32_f16(false, a[c].v,     false, be,
                                                          (short)0, acc0, false, false);
            acc1 = __builtin_amdgcn_wmma_f32_16x16x32_f16(false, a[c + 1].v, false, bo,
                                                          (short)0, acc1, false, false);
        }
        const v8f acc = acc0 + acc1;
        const float bias = b1[col];
        #pragma unroll
        for (int v = 0; v < 8; ++v) {
            const int rowt = v + rb;
            const float g = gelu_exact(acc[v] + bias);
            if (n < 8) {
                zloc[(size_t)(r0 + rowt) * 128 + col] = (_Float16)g;
            } else {
                atomicAdd(&ssum[sseg[wave][rowt] * 128 + (n - 8) * 16 + ncol], g);
            }
        }
    }

    __syncthreads();
    for (int i = tid; i < 16 * 128; i += 128) atomicAdd(sums + i, ssum[i]);
}

// ---------------- means: sums / max(count,1), counts = off[s+1]-off[s] ----------
__global__ void means_kernel(const float* __restrict__ sums,
                             const long long* __restrict__ boff,
                             _Float16* __restrict__ means16) {
    int t = blockIdx.x * blockDim.x + threadIdx.x;
    if (t >= 16 * 128) return;
    int s = t >> 7;
    float cnt = (float)(boff[s + 1] - boff[s]);
    means16[t] = (_Float16)(sums[t] / fmaxf(cnt, 1.0f));
}

// ---------------- phase 2: [z_local|means] -> GEMM2/3 (WMMA) -> GEMM4 -> out ----
__global__ void __launch_bounds__(128, 1)
phase2_kernel(const long long* __restrict__ boff,
              const _Float16* __restrict__ zloc, const _Float16* __restrict__ means16,
              const _Float16* __restrict__ w2T, const float* __restrict__ b2,
              const _Float16* __restrict__ w3T, const float* __restrict__ b3,
              const float* __restrict__ w4, const float* __restrict__ b4,
              float* __restrict__ out) {
    __shared__ _Float16 sA [4][16 * 256];  // z2 tile, 8KB/wave
    __shared__ _Float16 sH2[4][16 * 128];  // gelu(GEMM2) tile, 4KB/wave
    __shared__ _Float16 sH3[4][16 * 64];   // gelu(GEMM3) tile, 2KB/wave
    __shared__ int      sseg[4][16];

    const int tid  = threadIdx.x;
    const int wave = tid >> 5;
    const int lane = tid & 31;
    const bool lo16 = (lane < 16);
    const int r0   = (blockIdx.x * 4 + wave) * 16;

    if (lo16) {
        long long row = r0 + lane;
        int s = 0;
        #pragma unroll
        for (int j = 1; j <= 16; ++j) s += (boff[j] <= row) ? 1 : 0;
        sseg[wave][lane] = s;
    }

    // build A2 = [z_local(16x128) | means[seg](16x128)] in LDS
    {
        const int rhalf = lane >> 4;          // 0/1
        const int col8  = (lane & 15) * 8;    // 0..120
        #pragma unroll
        for (int it = 0; it < 8; ++it) {
            const int r = it * 2 + rhalf;
            *(half8*)&sA[wave][r * 256 + col8] =
                *(const half8*)(zloc + (size_t)(r0 + r) * 128 + col8);
            const int s = sseg[wave][r];
            *(half8*)&sA[wave][r * 256 + 128 + col8] =
                *(const half8*)(means16 + s * 128 + col8);
        }
    }

    const int arow = lane & 15;
    const int koff = lo16 ? 0 : 16;
    const int ncol = lane & 15;
    const int kb2  = lo16 ? 0 : 16;
    const int rb   = lo16 ? 0 : 8;

    // ---- GEMM2: 16x128, K=256 ----
    {
        AFrag a[8];
        #pragma unroll
        for (int c = 0; c < 8; ++c) {
            const _Float16* p = &sA[wave][arow * 256 + c * 32 + koff];
            a[c].h[0] = *(const half8*)p;
            a[c].h[1] = *(const half8*)(p + 8);
        }
        for (int n = 0; n < 8; ++n) {
            const int col = n * 16 + ncol;
            const _Float16* bp = w2T + col * 256 + kb2;
            v8f acc0 = {};
            v8f acc1 = {};
            #pragma unroll
            for (int c = 0; c < 8; c += 2) {
                v16h be = *(const v16h*)(bp + c * 32);
                v16h bo = *(const v16h*)(bp + c * 32 + 32);
                acc0 = __builtin_amdgcn_wmma_f32_16x16x32_f16(false, a[c].v,     false, be,
                                                              (short)0, acc0, false, false);
                acc1 = __builtin_amdgcn_wmma_f32_16x16x32_f16(false, a[c + 1].v, false, bo,
                                                              (short)0, acc1, false, false);
            }
            const v8f acc = acc0 + acc1;
            const float bias = b2[col];
            #pragma unroll
            for (int v = 0; v < 8; ++v)
                sH2[wave][(v + rb) * 128 + col] = (_Float16)gelu_exact(acc[v] + bias);
        }
    }

    // ---- GEMM3: 16x64, K=128 ----
    {
        AFrag a[4];
        #pragma unroll
        for (int c = 0; c < 4; ++c) {
            const _Float16* p = &sH2[wave][arow * 128 + c * 32 + koff];
            a[c].h[0] = *(const half8*)p;
            a[c].h[1] = *(const half8*)(p + 8);
        }
        for (int n = 0; n < 4; ++n) {
            const int col = n * 16 + ncol;
            const _Float16* bp = w3T + col * 128 + kb2;
            v8f acc0 = {};
            v8f acc1 = {};
            #pragma unroll
            for (int c = 0; c < 4; c += 2) {
                v16h be = *(const v16h*)(bp + c * 32);
                v16h bo = *(const v16h*)(bp + c * 32 + 32);
                acc0 = __builtin_amdgcn_wmma_f32_16x16x32_f16(false, a[c].v,     false, be,
                                                              (short)0, acc0, false, false);
                acc1 = __builtin_amdgcn_wmma_f32_16x16x32_f16(false, a[c + 1].v, false, bo,
                                                              (short)0, acc1, false, false);
            }
            const v8f acc = acc0 + acc1;
            const float bias = b3[col];
            #pragma unroll
            for (int v = 0; v < 8; ++v)
                sH3[wave][(v + rb) * 64 + col] = (_Float16)gelu_exact(acc[v] + bias);
        }
    }

    // ---- GEMM4: 16x64 @ 64x1 via split-lane dot, half-wave reduce ----
    {
        const int r  = lane & 15;
        const int cb = lo16 ? 0 : 32;
        float acc = 0.0f;
        #pragma unroll
        for (int i = 0; i < 32; ++i)
            acc += (float)sH3[wave][r * 64 + cb + i] * w4[cb + i];
        acc += __shfl_xor(acc, 16, 32);
        if (lo16) out[r0 + r] = acc + b4[0];
    }
}

// ---------------- launch ----------------
extern "C" void kernel_launch(void* const* d_in, const int* in_sizes, int n_in,
                              void* d_out, int out_size, void* d_ws, size_t ws_size,
                              hipStream_t stream) {
    const float*     x    = (const float*)d_in[0];
    const long long* boff = (const long long*)d_in[1];
    const float*     lnw  = (const float*)d_in[2];
    const float*     lnb  = (const float*)d_in[3];
    const float*     w1   = (const float*)d_in[4];
    const float*     b1   = (const float*)d_in[5];
    const float*     w2   = (const float*)d_in[6];
    const float*     b2   = (const float*)d_in[7];
    const float*     w3   = (const float*)d_in[8];
    const float*     b3   = (const float*)d_in[9];
    const float*     w4   = (const float*)d_in[10];
    const float*     b4   = (const float*)d_in[11];
    float* out = (float*)d_out;

    char* ws = (char*)d_ws;
    _Float16* w1T     = (_Float16*)(ws + OFF_W1T);
    _Float16* w2T     = (_Float16*)(ws + OFF_W2T);
    _Float16* w3T     = (_Float16*)(ws + OFF_W3T);
    float*    sums    = (float*)   (ws + OFF_SUMS);
    _Float16* means16 = (_Float16*)(ws + OFF_MEANS);
    _Float16* zloc    = (_Float16*)(ws + OFF_ZLOC);

    prep_kernel  <<<256, 256, 0, stream>>>(w1, w2, w3, w1T, w2T, w3T, sums);
    phase1_kernel<<<NROWS / 64, 128, 0, stream>>>(x, boff, lnw, lnb, w1T, b1, zloc, sums);
    means_kernel <<<8, 256, 0, stream>>>(sums, boff, means16);
    phase2_kernel<<<NROWS / 64, 128, 0, stream>>>(boff, zloc, means16, w2T, b2, w3T, b3,
                                                  w4, b4, out);
}
